// QuantizationLayer_37366215475690
// MI455X (gfx1250) — matrix operand order repaired and checked by reading, not compiled, not run on hardware
//
#include <hip/hip_runtime.h>

// ---------------------------------------------------------------------------
// VQ quantization: N=16384 rows, D=512 dims, K=8192 codes.
//   scores[n,k] = ||w_k||^2 - 2 <x_n, w_k>   (||x_n||^2 irrelevant to argmin)
//   idx[n] = argmin_k scores ; out = W[idx] ; loss = 1.25 * mean((out-x)^2)
// GEMM: v_wmma_f32_16x16x32_bf16, fp32 accumulate, argmin fused in epilogue.
// Data movement: one-time fp32->bf16 conversion to scratch, then
// global_load_async_to_lds_b128 (ASYNCcnt) double-buffered W staging; the
// x-tile is async-loaded once and stays LDS-resident. The inner loop is
// software-pipelined: fragments for K-step kk+1 are fetched while the four
// WMMAs of step kk execute (avoids dscnt==0 serialization).
// ---------------------------------------------------------------------------

#define N_ROWS  16384
#define D_DIM   512
#define K_CODES 8192

#define BM 64            // x rows per workgroup (LDS-resident)
#define BN 128           // codebook rows per chunk
#define BK 128           // d-slice of W staged per async chunk
#define XSTR (D_DIM + 8) // padded bf16 row stride for x tile
#define WSTR (BK + 8)    // padded bf16 row stride for W chunk

typedef __attribute__((ext_vector_type(8)))  float  v8f;
typedef __attribute__((ext_vector_type(8)))  __bf16 v8bf;
typedef __attribute__((ext_vector_type(16))) __bf16 v16bf;

__device__ __forceinline__ unsigned short f2bf(float f) {
  unsigned int u = __builtin_bit_cast(unsigned int, f);
  u += 0x7FFFu + ((u >> 16) & 1u);            // round-to-nearest-even
  return (unsigned short)(u >> 16);
}

// gfx1250 async global->LDS copy, 16B per lane, GVS mode (SGPR base + V offset)
__device__ __forceinline__ void async_cp16(unsigned lds_off, unsigned goff,
                                           const void* base) {
  asm volatile("global_load_async_to_lds_b128 %0, %1, %2"
               :: "v"(lds_off), "v"(goff), "s"(base) : "memory");
}
#define WAIT_ASYNC(n) asm volatile("s_wait_asynccnt %0" :: "n"(n) : "memory")

__device__ __forceinline__ v16bf ld_frag(const unsigned short* p0,
                                         const unsigned short* p1) {
  v8bf lo = *(const v8bf*)p0;
  v8bf hi = *(const v8bf*)p1;
  return __builtin_shufflevector(lo, hi,
      0,1,2,3,4,5,6,7,8,9,10,11,12,13,14,15);
}

// --------------------------------------------------------- fp32 -> bf16 copy
__global__ __launch_bounds__(256) void cvt_bf16_kernel(const float* __restrict__ src,
                                                       unsigned short* __restrict__ dst) {
  const int i = blockIdx.x * 256 + threadIdx.x;      // one float4 per thread
  float4 v = *(const float4*)(src + (size_t)i * 4);
  ushort4 u;
  u.x = f2bf(v.x); u.y = f2bf(v.y); u.z = f2bf(v.z); u.w = f2bf(v.w);
  *(ushort4*)(dst + (size_t)i * 4) = u;
}

// -------------------------------------------------------------------- wnorm
__global__ __launch_bounds__(256) void wnorm_kernel(const float* __restrict__ W,
                                                    float* __restrict__ wnorm) {
  const int row  = blockIdx.x * 8 + (threadIdx.x >> 5);
  const int lane = threadIdx.x & 31;
  const float* wr = W + (size_t)row * D_DIM;
  float s = 0.0f;
#pragma unroll
  for (int i = 0; i < D_DIM / 32; ++i) {
    float v = wr[lane + i * 32];
    s = fmaf(v, v, s);
  }
#pragma unroll
  for (int off = 16; off >= 1; off >>= 1) s += __shfl_xor(s, off, 32);
  if (lane == 0) wnorm[row] = s;
}

// ------------------------------------------------- fused GEMM + argmin (WMMA)
__global__ __launch_bounds__(256) void vq_argmin_kernel(
    const unsigned short* __restrict__ xb,   // bf16 x   [N_ROWS, D_DIM]
    const unsigned short* __restrict__ wb,   // bf16 W   [K_CODES, D_DIM]
    const float* __restrict__ wnorm,
    int* __restrict__ idxOut) {
  __shared__ __attribute__((aligned(16))) unsigned short xs[BM * XSTR];
  __shared__ __attribute__((aligned(16))) unsigned short wbuf[2][BN * WSTR];
  __shared__ float fbv[BM][4];
  __shared__ int   fbi[BM][4];

  const int tid  = threadIdx.x;
  const int lane = tid & 31;
  const int wave = tid >> 5;
  const int bm   = blockIdx.x * BM;
  const int mw   = (wave & 1) * 32;   // wave's 32-row M subtile
  const int nq   = wave >> 1;         // wave's 32-col N quarter (0..3)
  const int half = lane >> 4;         // 0: lanes 0-15, 1: lanes 16-31
  const int l16  = lane & 15;

  const unsigned xs_base = (unsigned)(unsigned long long)&xs[0];
  const unsigned wb_base = (unsigned)(unsigned long long)&wbuf[0][0];
  const unsigned WBUF_B  = (unsigned)(BN * WSTR * 2);

  // ---- async-load the whole x tile (BM x D, bf16) once: 16 ops/thread -----
#pragma unroll
  for (int t = 0; t < 16; ++t) {
    int lin = tid + t * 256;                 // 0..4095
    int row = lin >> 6;                      // 64 transfers per row
    int c8  = (lin & 63) << 3;               // 8 bf16 per transfer
    async_cp16(xs_base + (unsigned)(row * XSTR + c8) * 2,
               (unsigned)((bm + row) * D_DIM + c8) * 2, xb);
  }
  // ---- async-load W chunk 0 (rows 0..127, d 0..127): 8 ops/thread ---------
#pragma unroll
  for (int t = 0; t < 8; ++t) {
    int lin = tid + t * 256;                 // 0..2047
    int row = lin >> 4;                      // 16 transfers per row
    int c8  = (lin & 15) << 3;
    async_cp16(wb_base + (unsigned)(row * WSTR + c8) * 2,
               (unsigned)(row * D_DIM + c8) * 2, wb);
  }

  float best [2][8];
  int   besti[2][8];
#pragma unroll
  for (int a = 0; a < 2; ++a)
#pragma unroll
    for (int r = 0; r < 8; ++r) { best[a][r] = 3.4e38f; besti[a][r] = 0; }

  v8f acc[2][2];                         // [a: M half 16][t: N half 16]

  const int arow0 = (mw + l16) * XSTR;
  const int arow1 = (mw + 16 + l16) * XSTR;
  const int brow0 = (nq * 32 + l16) * WSTR;
  const int brow1 = (nq * 32 + 16 + l16) * WSTR;

  // flat chunk index: nb = c>>2 (128-col code block), db = (c&3)*128 (d slice)
  for (int c = 0; c < (K_CODES / BN) * (D_DIM / BK); ++c) {
    if (c < (K_CODES / BN) * (D_DIM / BK) - 1) {
      // prefetch next W chunk into the other buffer
      const int cn = c + 1;
      const int nb = (cn >> 2) * BN;
      const int db = (cn & 3) * BK;
      const unsigned bufo = (unsigned)(cn & 1) * WBUF_B;
#pragma unroll
      for (int t = 0; t < 8; ++t) {
        int lin = tid + t * 256;
        int row = lin >> 4;
        int c8  = (lin & 15) << 3;
        async_cp16(wb_base + bufo + (unsigned)(row * WSTR + c8) * 2,
                   (unsigned)((nb + row) * D_DIM + db + c8) * 2, wb);
      }
      WAIT_ASYNC(8);                         // current chunk landed
    } else {
      WAIT_ASYNC(0);
    }
    __syncthreads();

    if ((c & 3) == 0) {
#pragma unroll
      for (int a = 0; a < 2; ++a)
#pragma unroll
        for (int t = 0; t < 2; ++t)
#pragma unroll
          for (int e = 0; e < 8; ++e) acc[a][t][e] = 0.0f;
    }

    const unsigned short* wcur = &wbuf[c & 1][0];
    const int dxb = (c & 3) << 7;            // d-offset of this chunk in x tile

    // ---- software-pipelined K loop: load step kk+1 during WMMAs of kk -----
    v16bf A0[2], B0[2];
    {
      const int ak = dxb + (half ? 8 : 0);
      const int bk = (half ? 8 : 0);
      A0[0] = ld_frag(&xs[arow0 + ak], &xs[arow0 + ak + 16]);
      A0[1] = ld_frag(&xs[arow1 + ak], &xs[arow1 + ak + 16]);
      B0[0] = ld_frag(&wcur[brow0 + bk], &wcur[brow0 + bk + 16]);
      B0[1] = ld_frag(&wcur[brow1 + bk], &wcur[brow1 + bk + 16]);
    }
#pragma unroll
    for (int kk = 0; kk < BK / 32; ++kk) {
      v16bf A1[2], B1[2];
      if (kk < BK / 32 - 1) {
        const int ak = dxb + (kk + 1) * 32 + (half ? 8 : 0);
        const int bk = (kk + 1) * 32 + (half ? 8 : 0);
        A1[0] = ld_frag(&xs[arow0 + ak], &xs[arow0 + ak + 16]);
        A1[1] = ld_frag(&xs[arow1 + ak], &xs[arow1 + ak + 16]);
        B1[0] = ld_frag(&wcur[brow0 + bk], &wcur[brow0 + bk + 16]);
        B1[1] = ld_frag(&wcur[brow1 + bk], &wcur[brow1 + bk + 16]);
      }
      acc[0][0] = __builtin_amdgcn_wmma_f32_16x16x32_bf16(
          false, A0[0], false, B0[0], (short)0, acc[0][0], false, false);
      acc[1][0] = __builtin_amdgcn_wmma_f32_16x16x32_bf16(
          false, A0[1], false, B0[0], (short)0, acc[1][0], false, false);
      acc[0][1] = __builtin_amdgcn_wmma_f32_16x16x32_bf16(
          false, A0[0], false, B0[1], (short)0, acc[0][1], false, false);
      acc[1][1] = __builtin_amdgcn_wmma_f32_16x16x32_bf16(
          false, A0[1], false, B0[1], (short)0, acc[1][1], false, false);
      if (kk < BK / 32 - 1) {
        A0[0] = A1[0]; A0[1] = A1[1];
        B0[0] = B1[0]; B0[1] = B1[1];
      }
    }

    if ((c & 3) == 3) {
      // Epilogue for this 128-column code block: score + running argmin.
      // C layout: VGPR r, lanes 0-15 -> (M=r, N=lane); lanes 16-31 -> M=r+8.
      const int nbase = (c >> 2) * BN;
#pragma unroll
      for (int t = 0; t < 2; ++t) {
        const int nglob = nbase + nq * 32 + t * 16 + l16;
        const float wn = wnorm[nglob];
#pragma unroll
        for (int a = 0; a < 2; ++a)
#pragma unroll
          for (int r = 0; r < 8; ++r) {
            float score = fmaf(-2.0f, acc[a][t][r], wn);
            if (score < best[a][r]) { best[a][r] = score; besti[a][r] = nglob; }
          }
      }
    }
    __syncthreads();
  }

  // Cross-lane (val,idx) min within each 16-lane half (wave32).
#pragma unroll
  for (int a = 0; a < 2; ++a)
#pragma unroll
    for (int r = 0; r < 8; ++r) {
      float bv = best[a][r];
      int   bi = besti[a][r];
#pragma unroll
      for (int off = 8; off >= 1; off >>= 1) {
        float ov = __shfl_xor(bv, off, 32);
        int   oi = __shfl_xor(bi, off, 32);
        if (ov < bv || (ov == bv && oi < bi)) { bv = ov; bi = oi; }
      }
      best[a][r] = bv; besti[a][r] = bi;
    }
  if (l16 == 0) {
#pragma unroll
    for (int a = 0; a < 2; ++a) {
      const int rb = mw + a * 16 + (half ? 8 : 0);
#pragma unroll
      for (int r = 0; r < 8; ++r) {
        fbv[rb + r][nq] = best[a][r];
        fbi[rb + r][nq] = besti[a][r];
      }
    }
  }
  __syncthreads();
  // combine the four N-quarters per row
  if (tid < BM) {
    float bv = fbv[tid][0];
    int   bi = fbi[tid][0];
#pragma unroll
    for (int q = 1; q < 4; ++q) {
      float ov = fbv[tid][q];
      int   oi = fbi[tid][q];
      if (ov < bv || (ov == bv && oi < bi)) { bv = ov; bi = oi; }
    }
    idxOut[bm + tid] = bi;
  }
}

// ----------------------------------------------- gather + per-row sq error
__global__ __launch_bounds__(256) void gather_loss_kernel(const float* __restrict__ x,
                                                          const float* __restrict__ W,
                                                          const int* __restrict__ idx,
                                                          float* __restrict__ out,
                                                          float* __restrict__ partial) {
  const int row = blockIdx.x;
  const int k   = idx[row];
  const int d0  = threadIdx.x * 2;
  float2 q  = *(const float2*)(W + (size_t)k   * D_DIM + d0);   // original fp32 W
  float2 xv = *(const float2*)(x + (size_t)row * D_DIM + d0);
  *(float2*)(out + (size_t)row * D_DIM + d0) = q;   // straight-through fwd value
  float dx = q.x - xv.x, dy = q.y - xv.y;
  float s = dx * dx + dy * dy;
#pragma unroll
  for (int off = 16; off >= 1; off >>= 1) s += __shfl_xor(s, off, 32);
  __shared__ float red[8];
  const int lane = threadIdx.x & 31, wv = threadIdx.x >> 5;
  if (lane == 0) red[wv] = s;
  __syncthreads();
  if (threadIdx.x == 0) {
    float tot = 0.0f;
#pragma unroll
    for (int i = 0; i < 8; ++i) tot += red[i];
    partial[row] = tot;     // deterministic: no atomics
  }
}

// -------------------------------------------------------------- final scalar
__global__ __launch_bounds__(256) void finalize_kernel(const float* __restrict__ partial,
                                                       float* __restrict__ out) {
  __shared__ float red[256];
  float s = 0.0f;
  for (int i = threadIdx.x; i < N_ROWS; i += 256) s += partial[i];
  red[threadIdx.x] = s;
  __syncthreads();
  for (int st = 128; st > 0; st >>= 1) {
    if (threadIdx.x < st) red[threadIdx.x] += red[threadIdx.x + st];
    __syncthreads();
  }
  if (threadIdx.x == 0)
    out[(size_t)N_ROWS * D_DIM] =
        1.25f * red[0] * (1.0f / (float)((size_t)N_ROWS * D_DIM));
}

// ---------------------------------------------------------------------------
extern "C" void kernel_launch(void* const* d_in, const int* in_sizes, int n_in,
                              void* d_out, int out_size, void* d_ws, size_t ws_size,
                              hipStream_t stream) {
  (void)in_sizes; (void)n_in; (void)out_size; (void)ws_size;
  const float* x = (const float*)d_in[0];   // [16384, 512] f32
  const float* W = (const float*)d_in[1];   // [8192, 512] f32
  float* out = (float*)d_out;               // [16384*512 + 1] f32

  // Workspace layout (all fully rewritten every call):
  //   [64, +64K)   : idx, 16384 x int
  //   [.., +32K)   : wnorm, 8192 x float
  //   [.., +64K)   : partial, 16384 x float
  //   [256K, +16M) : x in bf16
  //   [.., +8M)    : W in bf16
  char* ws = (char*)d_ws;
  int*            idx     = (int*)  (ws + 64);
  float*          wnorm   = (float*)(ws + 64 + N_ROWS * sizeof(int));
  float*          partial = (float*)(ws + 64 + N_ROWS * sizeof(int) + K_CODES * sizeof(float));
  unsigned short* xb      = (unsigned short*)(ws + (1 << 18));
  unsigned short* wb      = (unsigned short*)(ws + (1 << 18) + (size_t)N_ROWS * D_DIM * 2);

  cvt_bf16_kernel  <<<(N_ROWS  * D_DIM / 4) / 256, 256, 0, stream>>>(x, xb);
  cvt_bf16_kernel  <<<(K_CODES * D_DIM / 4) / 256, 256, 0, stream>>>(W, wb);
  wnorm_kernel     <<<K_CODES / 8, 256, 0, stream>>>(W, wnorm);
  vq_argmin_kernel <<<N_ROWS / BM, 256, 0, stream>>>(xb, wb, wnorm, idx);
  gather_loss_kernel<<<N_ROWS,     256, 0, stream>>>(x, W, idx, out, partial);
  finalize_kernel  <<<1,           256, 0, stream>>>(partial, out);
}
